// HybridGNN_48799418417574
// MI455X (gfx1250) — compile-verified, matrix operand order checked by reading, not run on hardware
//
#include <hip/hip_runtime.h>
#include <math.h>

// ---------------- problem constants (from reference) ----------------
#define NN 20000
#define EE 160000
#define NG 20
#define GN_EPS 1e-5f

// ---------------- WMMA types ----------------
typedef __bf16  bf16_t;
typedef bf16_t  v16bf __attribute__((ext_vector_type(16)));
typedef float   v8f   __attribute__((ext_vector_type(8)));

__device__ __forceinline__ unsigned short f2bf(float f) {
  unsigned u = __float_as_uint(f);
  u += 0x7FFFu + ((u >> 16) & 1u);          // round-to-nearest-even
  return (unsigned short)(u >> 16);
}

__device__ __forceinline__ void atomicMaxF(float* addr, float v) {
  // sign-aware int trick; buffer initialized to -inf bits (0xFF800000)
  if (v >= 0.0f) atomicMax((int*)addr, __float_as_int(v));
  else           atomicMin((unsigned int*)addr, __float_as_uint(v));
}

// =====================================================================
// Segmented gather-GEMM:  C[M,N] = act( A[M,K] @ W[K,N] + bias )
//   A is composed of up to 3 horizontal segments (each segK wide) with
//   optional per-row gather indices (h[dst] / h[src] concatenation).
//   f32 in memory, bf16 in LDS, f32 WMMA accumulation.
// mode 0: store acc+bias
// mode 1: attention epilogue: leaky_relu(acc+bias,0.1) * w2[col],
//         reduce over columns, atomicAdd into attOut[row]  (att_w2 fused)
// mode 2: store relu(acc+bias)
// Block: 256 thr = 8 waves; tile 64(M) x 128(N), BK=64.
// Wave w: rows (w>>1)*16, cols (w&1)*64 -> 4 accumulators, 8 WMMA/stage.
// =====================================================================
__global__ __launch_bounds__(256)
void k_gemm(const float* __restrict__ A0, const float* __restrict__ A1,
            const float* __restrict__ A2,
            const int* __restrict__ idx0, const int* __restrict__ idx1,
            const int* __restrict__ idx2,
            int segK, const float* __restrict__ Wm, const float* __restrict__ bias,
            int M, int K, int N, int mode,
            float* __restrict__ C, const float* __restrict__ w2,
            float* __restrict__ attOut)
{
  __shared__ __attribute__((aligned(16))) unsigned short lsA[64][68];
  __shared__ __attribute__((aligned(16))) unsigned short lsB[128][68];

  const int t    = threadIdx.x;
  const int m0   = blockIdx.y * 64;
  const int n0   = blockIdx.x * 128;
  const int wave = t >> 5, lane = t & 31;
  const int wm   = (wave >> 1) * 16;   // wave row offset in tile
  const int wn   = (wave & 1) * 64;    // wave col offset in tile
  const int lhalf = lane >> 4, l15 = lane & 15;
  const int kb   = lhalf * 8;

  // ---- per-thread A staging assignment: one row, 16 contiguous k
  const int arow   = t >> 2;           // 0..63
  const int kchunk = (t & 3) * 16;     // 0/16/32/48
  const int gm_s   = m0 + arow;
  // pre-gather segment row indices ONCE (hoisted out of the K loop)
  int r0 = 0, r1 = 0, r2 = 0;
  if (gm_s < M) {
    r0 = idx0 ? idx0[gm_s] : gm_s;
    r1 = idx1 ? idx1[gm_s] : gm_s;
    r2 = idx2 ? idx2[gm_s] : gm_s;
  }
  const int segK2 = segK * 2;

  v8f acc[4] = {};

  for (int k0 = 0; k0 < K; k0 += 64) {
    // prefetch next weight tile through L2 (global_prefetch_b8)
    if (k0 + 64 < K)
      __builtin_prefetch(&Wm[(size_t)(k0 + 64) * N + n0 + (t & 127)], 0, 1);

    // ---- stage A tile [64][64] (gather + f32->bf16)
#pragma unroll
    for (int i = 0; i < 16; ++i) {
      int k  = kchunk + i;
      int gk = k0 + k;
      float v = 0.0f;
      if (gm_s < M && gk < K) {
        int seg = (gk >= segK) + (gk >= segK2);   // no integer division
        int kk  = gk - seg * segK;
        const float* Ap = (seg == 0) ? A0 : ((seg == 1) ? A1 : A2);
        int row = (seg == 0) ? r0 : ((seg == 1) ? r1 : r2);
        v = Ap[(size_t)row * segK + kk];
      }
      lsA[arow][k] = f2bf(v);
    }
    // ---- stage B tile transposed to [n][k]
    for (int i = 0; i < 32; ++i) {
      int e  = t + i * 256;
      int k  = e >> 7, n = e & 127;
      int gk = k0 + k, gn = n0 + n;
      float v = (gk < K && gn < N) ? Wm[(size_t)gk * N + gn] : 0.0f;
      lsB[n][k] = f2bf(v);
    }
    __syncthreads();

    // ---- 2 K-steps x 4 column tiles = 8 WMMAs per wave per stage
#pragma unroll
    for (int ks = 0; ks < 2; ++ks) {
      const int kbase = ks * 32 + kb;
      union { v16bf v; unsigned u[8]; } fa;
#pragma unroll
      for (int j = 0; j < 8; ++j) {
        int kk = kbase + ((j < 4) ? (2 * j) : (16 + 2 * (j - 4)));
        fa.u[j] = *(const unsigned*)&lsA[wm + l15][kk];
      }
#pragma unroll
      for (int tc = 0; tc < 4; ++tc) {
        union { v16bf v; unsigned u[8]; } fb;
#pragma unroll
        for (int j = 0; j < 8; ++j) {
          int kk = kbase + ((j < 4) ? (2 * j) : (16 + 2 * (j - 4)));
          fb.u[j] = *(const unsigned*)&lsB[wn + tc * 16 + l15][kk];
        }
        acc[tc] = __builtin_amdgcn_wmma_f32_16x16x32_bf16(false, fa.v, false, fb.v,
                                                          (short)0, acc[tc], false, false);
      }
    }
    __syncthreads();
  }

  if (mode == 1) {
    // fused att_w2 projection: per-row dot over this block's 128 columns
    float bcol[4], scol[4];
#pragma unroll
    for (int tc = 0; tc < 4; ++tc) {
      int col = n0 + wn + tc * 16 + l15;
      bcol[tc] = (col < N) ? bias[col] : 0.0f;
      scol[tc] = (col < N) ? w2[col]  : 0.0f;
    }
    float rowp[8];
#pragma unroll
    for (int r = 0; r < 8; ++r) {
      float s = 0.0f;
#pragma unroll
      for (int tc = 0; tc < 4; ++tc) {
        float a = acc[tc][r] + bcol[tc];
        a = (a > 0.0f) ? a : 0.1f * a;          // leaky_relu 0.1
        s += a * scol[tc];
      }
      rowp[r] = s;
    }
#pragma unroll
    for (int msk = 1; msk < 16; msk <<= 1) {
#pragma unroll
      for (int r = 0; r < 8; ++r) rowp[r] += __shfl_xor(rowp[r], msk, 32);
    }
    if (l15 == 0) {
#pragma unroll
      for (int r = 0; r < 8; ++r) {
        int gm = m0 + wm + lhalf * 8 + r;
        if (gm < M) atomicAdd(&attOut[gm], rowp[r]);
      }
    }
  } else {
#pragma unroll
    for (int tc = 0; tc < 4; ++tc) {
      int col = n0 + wn + tc * 16 + l15;
      if (col >= N) continue;
      float b = bias[col];
#pragma unroll
      for (int r = 0; r < 8; ++r) {
        int gm = m0 + wm + lhalf * 8 + r;
        if (gm >= M) continue;
        float v = acc[tc][r] + b;
        if (mode == 2) v = fmaxf(v, 0.0f);
        C[(size_t)gm * N + col] = v;
      }
    }
  }
}

// ===================== elementwise / scatter kernels =====================

__global__ void k_node_init(float* m, float* z, float* cnt, float* sums,
                            float* maxb, float* attout, int Nn, int D) {
  int i = blockIdx.x * blockDim.x + threadIdx.x;
  if (i >= Nn * D) return;
  sums[i]   = 0.0f;
  maxb[i]   = __int_as_float(0xFF800000);   // -inf
  attout[i] = 0.0f;
  if ((i % D) == 0) {
    int n = i / D;
    m[n] = __int_as_float(0xFF800000);
    z[n] = 0.0f; cnt[n] = 0.0f;
  }
}

__global__ void k_edge_init(float* att, const float* b2, int E) {
  int e = blockIdx.x * blockDim.x + threadIdx.x;
  if (e < E) att[e] = b2[0];
}

__global__ void k_att_max(const float* att, const int* dst, float* m, int E) {
  int e = blockIdx.x * blockDim.x + threadIdx.x;
  if (e < E) atomicMaxF(&m[dst[e]], att[e]);
}

__global__ void k_att_exp(const float* att, const int* dst, const float* m,
                          float* ew, float* z, float* cnt, int E) {
  int e = blockIdx.x * blockDim.x + threadIdx.x;
  if (e >= E) return;
  int d = dst[e];
  float v = expf(att[e] - m[d]);
  ew[e] = v;
  atomicAdd(&z[d], v);
  atomicAdd(&cnt[d], 1.0f);
}

// triple scatter-reduce: sum / max / softmax-weighted sum, one block per edge
__global__ void k_scatter(const float* __restrict__ h, const int* __restrict__ src,
                          const int* __restrict__ dst, const float* __restrict__ ew,
                          const float* __restrict__ z,
                          float* sums, float* maxb, float* attout, int D) {
  int e = blockIdx.x;
  int s = src[e], d = dst[e];
  float w = ew[e] / z[d];
  size_t rs = (size_t)s * D, rd = (size_t)d * D;
  for (int c = threadIdx.x; c < D; c += blockDim.x) {
    float hj = h[rs + c];
    atomicAdd(&sums[rd + c], hj);
    atomicMaxF(&maxb[rd + c], hj);
    atomicAdd(&attout[rd + c], w * hj);
  }
}

__global__ void k_finalize(float* sums /* -> mean */, float* maxb,
                           const float* cnt, int Nn, int D) {
  int i = blockIdx.x * blockDim.x + threadIdx.x;
  if (i >= Nn * D) return;
  float c = cnt[i / D];
  sums[i] = sums[i] / fmaxf(c, 1.0f);
  if (!(c > 0.0f)) maxb[i] = 0.0f;    // empty segments -> 0 (torch_scatter)
}

// gate layer-2 ([D]->3), softmax over 3, fuse aggregations, + residual h
__global__ void k_gate2fuse(const float* __restrict__ g, const float* __restrict__ w2,
                            const float* __restrict__ b2,
                            const float* __restrict__ meanb, const float* __restrict__ maxb,
                            const float* __restrict__ attb, const float* __restrict__ h,
                            float* __restrict__ out, int Nn, int D) {
  int w = (int)((blockIdx.x * blockDim.x + threadIdx.x) >> 5);  // 1 wave / node
  int lane = threadIdx.x & 31;
  if (w >= Nn) return;
  size_t row = (size_t)w * D;
  float d0 = 0.f, d1 = 0.f, d2 = 0.f;
  for (int c = lane; c < D; c += 32) {
    float gv = g[row + c];
    d0 += gv * w2[c * 3 + 0];
    d1 += gv * w2[c * 3 + 1];
    d2 += gv * w2[c * 3 + 2];
  }
#pragma unroll
  for (int m = 16; m >= 1; m >>= 1) {
    d0 += __shfl_xor(d0, m, 32);
    d1 += __shfl_xor(d1, m, 32);
    d2 += __shfl_xor(d2, m, 32);
  }
  d0 += b2[0]; d1 += b2[1]; d2 += b2[2];
  float mx = fmaxf(d0, fmaxf(d1, d2));
  float e0 = expf(d0 - mx), e1 = expf(d1 - mx), e2 = expf(d2 - mx);
  float inv = 1.0f / (e0 + e1 + e2);
  e0 *= inv; e1 *= inv; e2 *= inv;
  for (int c = lane; c < D; c += 32)
    out[row + c] = e0 * meanb[row + c] + e1 * maxb[row + c] + e2 * attb[row + c] + h[row + c];
}

// ===================== GraphNorm (+ fused relu) =====================

__global__ void k_zero(float* p, int n) {
  int i = blockIdx.x * blockDim.x + threadIdx.x;
  if (i < n) p[i] = 0.0f;
}

__global__ void k_gn_sum(const float* y, const int* batch, float* gsum,
                         float* gcnt, int Nn, int D) {
  int i = blockIdx.x * blockDim.x + threadIdx.x;
  if (i >= Nn * D) return;
  int n = i / D, d = i % D;
  int b = batch[n];
  atomicAdd(&gsum[b * D + d], y[i]);
  if (d == 0) atomicAdd(&gcnt[b], 1.0f);
}

__global__ void k_gn_center(float* y, const int* batch, const float* gsum,
                            const float* gcnt, const float* mscale,
                            float* gvar, int Nn, int D) {
  int i = blockIdx.x * blockDim.x + threadIdx.x;
  if (i >= Nn * D) return;
  int n = i / D, d = i % D;
  int b = batch[n];
  float c = fmaxf(gcnt[b], 1.0f);
  float xc = y[i] - mscale[d] * (gsum[b * D + d] / c);
  y[i] = xc;
  atomicAdd(&gvar[b * D + d], xc * xc);
}

__global__ void k_gn_norm(const float* y, const int* batch, const float* gvar,
                          const float* gcnt, const float* wt, const float* bs,
                          float* out, int Nn, int D) {
  int i = blockIdx.x * blockDim.x + threadIdx.x;
  if (i >= Nn * D) return;
  int n = i / D, d = i % D;
  int b = batch[n];
  float c = fmaxf(gcnt[b], 1.0f);
  float var = gvar[b * D + d] / c;
  float v = wt[d] * y[i] * rsqrtf(var + GN_EPS) + bs[d];
  out[i] = fmaxf(v, 0.0f);                   // relu fused (norm1..3 only)
}

// ===================== host orchestration =====================

extern "C" void kernel_launch(void* const* d_in, const int* in_sizes, int n_in,
                              void* d_out, int out_size, void* d_ws, size_t ws_size,
                              hipStream_t stream) {
  (void)in_sizes; (void)n_in; (void)out_size; (void)ws_size;
  // input order: x, conv1..conv4 (10 leaves each), norm1..norm3 (3 each),
  //              edge_index, batch
  const float* x     = (const float*)d_in[0];
  const int*   edge  = (const int*)d_in[50];
  const int*   srcI  = edge;            // edge_index[0]
  const int*   dstI  = edge + EE;       // edge_index[1]
  const int*   batch = (const int*)d_in[51];

  float* W = (float*)d_ws;
  size_t o = 0;
  auto take = [&](size_t nf) { float* p = W + o; o += nf; return p; };
  float* b_h      = take((size_t)NN * 512);
  float* b_actA   = take((size_t)NN * 512);
  float* b_actB   = take((size_t)NN * 512);
  float* b_y      = take((size_t)NN * 512);
  float* b_mean   = take((size_t)NN * 512);
  float* b_max    = take((size_t)NN * 512);
  float* b_attout = take((size_t)NN * 512);
  float* b_g      = take((size_t)NN * 512);
  float* b_att    = take((size_t)EE);
  float* b_ew     = take((size_t)EE);
  float* b_m      = take((size_t)NN);
  float* b_z      = take((size_t)NN);
  float* b_cnt    = take((size_t)NN);
  float* gsum     = take((size_t)NG * 512);
  float* gvar     = take((size_t)NG * 512);
  float* gcnt     = take((size_t)NG);

  const int din_[4]  = {50, 512, 512, 512};
  const int dout_[4] = {512, 512, 512, 121};
  const float* actIn = x;
  float* actBufs[2] = {b_actA, b_actB};

  for (int L = 0; L < 4; ++L) {
    const int din = din_[L], dout = dout_[L];
    const int pb = 1 + L * 10;
    const float* lin_w   = (const float*)d_in[pb + 0];
    const float* lin_b   = (const float*)d_in[pb + 1];
    const float* att_w1  = (const float*)d_in[pb + 2];
    const float* att_b1  = (const float*)d_in[pb + 3];
    const float* att_w2  = (const float*)d_in[pb + 4];
    const float* att_b2  = (const float*)d_in[pb + 5];
    const float* gate_w1 = (const float*)d_in[pb + 6];
    const float* gate_b1 = (const float*)d_in[pb + 7];
    const float* gate_w2 = (const float*)d_in[pb + 8];
    const float* gate_b2 = (const float*)d_in[pb + 9];

    // 1) h = x @ lin_w + lin_b   (WMMA)
    {
      dim3 g((dout + 127) / 128, (NN + 63) / 64);
      k_gemm<<<g, 256, 0, stream>>>(actIn, nullptr, nullptr, nullptr, nullptr, nullptr,
                                    din, lin_w, lin_b, NN, din, dout, 0,
                                    b_h, nullptr, nullptr);
    }
    // 2) init per-node / per-edge accumulators (att starts at b2)
    k_node_init<<<(NN * dout + 255) / 256, 256, 0, stream>>>(b_m, b_z, b_cnt,
                                                             b_mean, b_max, b_attout, NN, dout);
    k_edge_init<<<(EE + 255) / 256, 256, 0, stream>>>(b_att, att_b2, EE);
    // 3) att[e] += leaky_relu([h[dst],h[src]] @ att_w1 + b1) @ att_w2  (WMMA, fused)
    {
      dim3 g((dout + 127) / 128, (EE + 63) / 64);
      k_gemm<<<g, 256, 0, stream>>>(b_h, b_h, nullptr, dstI, srcI, nullptr,
                                    dout, att_w1, att_b1, EE, 2 * dout, dout, 1,
                                    nullptr, att_w2, b_att);
    }
    // 4) segment softmax over incoming edges
    k_att_max<<<(EE + 255) / 256, 256, 0, stream>>>(b_att, dstI, b_m, EE);
    k_att_exp<<<(EE + 255) / 256, 256, 0, stream>>>(b_att, dstI, b_m, b_ew, b_z, b_cnt, EE);
    // 5) triple scatter-reduce
    k_scatter<<<EE, 256, 0, stream>>>(b_h, srcI, dstI, b_ew, b_z,
                                      b_mean, b_max, b_attout, dout);
    k_finalize<<<(NN * dout + 255) / 256, 256, 0, stream>>>(b_mean, b_max, b_cnt, NN, dout);
    // 6) g = relu([mean,max,att] @ gate_w1 + b1)   (WMMA)
    {
      dim3 g((dout + 127) / 128, (NN + 63) / 64);
      k_gemm<<<g, 256, 0, stream>>>(b_mean, b_max, b_attout, nullptr, nullptr, nullptr,
                                    dout, gate_w1, gate_b1, NN, 3 * dout, dout, 2,
                                    b_g, nullptr, nullptr);
    }
    // 7) gate softmax(3) fusion + residual h
    float* yout = (L == 3) ? (float*)d_out : b_y;
    k_gate2fuse<<<(NN + 7) / 8, 256, 0, stream>>>(b_g, gate_w2, gate_b2,
                                                  b_mean, b_max, b_attout, b_h,
                                                  yout, NN, dout);
    // 8) GraphNorm + relu (layers 0..2)
    if (L < 3) {
      const int nb = 41 + L * 3;
      const float* nw  = (const float*)d_in[nb + 0];
      const float* nbs = (const float*)d_in[nb + 1];
      const float* nms = (const float*)d_in[nb + 2];
      k_zero<<<(NG * 512 + 255) / 256, 256, 0, stream>>>(gsum, NG * 512);
      k_zero<<<(NG * 512 + 255) / 256, 256, 0, stream>>>(gvar, NG * 512);
      k_zero<<<1, 32, 0, stream>>>(gcnt, NG);
      const int tot = NN * 512;
      k_gn_sum<<<(tot + 255) / 256, 256, 0, stream>>>(b_y, batch, gsum, gcnt, NN, 512);
      k_gn_center<<<(tot + 255) / 256, 256, 0, stream>>>(b_y, batch, gsum, gcnt, nms,
                                                         gvar, NN, 512);
      float* nxt = actBufs[L & 1];
      k_gn_norm<<<(tot + 255) / 256, 256, 0, stream>>>(b_y, batch, gvar, gcnt, nw, nbs,
                                                       nxt, NN, 512);
      actIn = nxt;
    }
  }
}
